// LATTE_23167053594632
// MI455X (gfx1250) — matrix operand order repaired and checked by reading, not compiled, not run on hardware
//
#include <hip/hip_runtime.h>
#include <hip/hip_bf16.h>

#define NN 50000
#define FF 256
#define DD 64
#define HH 4
#define EE 1000000
#define MM 2
#define HD 256   // H*D

typedef __attribute__((ext_vector_type(16))) __bf16 v16bf;
typedef __attribute__((ext_vector_type(8)))  __bf16 v8bf;
typedef __attribute__((ext_vector_type(8)))  float  v8f;

// ---------- helpers ----------
__device__ __forceinline__ unsigned fkey(float f) {
    unsigned u = __float_as_uint(f);
    return (u & 0x80000000u) ? ~u : (u | 0x80000000u);
}
__device__ __forceinline__ float finv(unsigned k) {
    if (k == 0u) return 0.0f;                       // empty-segment sentinel -> 0 (matches reference)
    unsigned u = (k & 0x80000000u) ? (k ^ 0x80000000u) : ~k;
    return __uint_as_float(u);
}

// ---------- prep: f32 -> bf16 hi/lo split ----------
__global__ void k_convert_x(const float* __restrict__ x, __bf16* __restrict__ xhi, __bf16* __restrict__ xlo) {
    int t = blockIdx.x * 256 + threadIdx.x;          // NN*FF threads exactly
    float v = x[t];
    __bf16 hi = (__bf16)v;
    float r = v - (float)hi;
    xhi[t] = hi;
    xlo[t] = (__bf16)r;
}

// Swizzle W[k][col] into B-operand order: idx = ((kstep*256 + col)*2 + khalf)*16 + j, k = kstep*32 + khalf*16 + j
__global__ void k_prep_w(const float* __restrict__ W, __bf16* __restrict__ Bhi, __bf16* __restrict__ Blo) {
    int t = blockIdx.x * 256 + threadIdx.x;          // 256*256 threads
    int k = t >> 8, col = t & 255;
    float v = W[t];
    __bf16 hi = (__bf16)v;
    float r = v - (float)hi;
    int kstep = k >> 5, half = (k >> 4) & 1, j = k & 15;
    int di = (((kstep * 256) + col) * 2 + half) * 16 + j;
    Bhi[di] = hi;
    Blo[di] = (__bf16)r;
}

// ---------- WMMA GEMM: h = x @ W + b  (bf16x3 split, f32 accumulate) ----------
__global__ void __launch_bounds__(256) k_gemm_h(const __bf16* __restrict__ xhi, const __bf16* __restrict__ xlo,
                                                const __bf16* __restrict__ Bhi, const __bf16* __restrict__ Blo,
                                                const float* __restrict__ bias, float* __restrict__ h) {
    int wave = (blockIdx.x << 3) + (threadIdx.x >> 5);   // 50000 tiles total
    int lane = threadIdx.x & 31;
    int tm = (wave >> 4) << 4;                           // row tile * 16
    int tn = (wave & 15) << 4;                           // col tile * 16
    int l15 = lane & 15, lh = lane >> 4;
    int arow = tm + l15;                                 // A: M = lane&15
    int col  = tn + l15;                                 // B/C: N = lane&15
    const __bf16* xh = xhi + (size_t)arow * FF + lh * 16; // A k-offset: 16*(lane>>4)
    const __bf16* xl = xlo + (size_t)arow * FF + lh * 16;
    v8f acc = {};
#pragma unroll
    for (int k0 = 0; k0 < FF; k0 += 32) {
        v8bf ah0 = *(const v8bf*)(xh + k0);
        v8bf ah1 = *(const v8bf*)(xh + k0 + 8);
        v8bf al0 = *(const v8bf*)(xl + k0);
        v8bf al1 = *(const v8bf*)(xl + k0 + 8);
        v16bf Ah, Al;
#pragma unroll
        for (int j = 0; j < 8; ++j) { Ah[j] = ah0[j]; Ah[j + 8] = ah1[j]; Al[j] = al0[j]; Al[j + 8] = al1[j]; }
        size_t bb = ((((size_t)(k0 >> 5) * 256 + col) * 2) + lh) * 16;
        v8bf bh0 = *(const v8bf*)(Bhi + bb);
        v8bf bh1 = *(const v8bf*)(Bhi + bb + 8);
        v8bf bl0 = *(const v8bf*)(Blo + bb);
        v8bf bl1 = *(const v8bf*)(Blo + bb + 8);
        v16bf Bh, Bl;
#pragma unroll
        for (int j = 0; j < 8; ++j) { Bh[j] = bh0[j]; Bh[j + 8] = bh1[j]; Bl[j] = bl0[j]; Bl[j + 8] = bl1[j]; }
        acc = __builtin_amdgcn_wmma_f32_16x16x32_bf16(false, Ah, false, Bh, (short)0, acc, false, false);
        acc = __builtin_amdgcn_wmma_f32_16x16x32_bf16(false, Ah, false, Bl, (short)0, acc, false, false);
        acc = __builtin_amdgcn_wmma_f32_16x16x32_bf16(false, Al, false, Bh, (short)0, acc, false, false);
    }
    float bc = bias[col];
#pragma unroll
    for (int r = 0; r < 8; ++r) {                        // C: M = r + 8*(lane>>4)
        int m = tm + (lh << 3) + r;
        h[(size_t)m * HD + col] = acc[r] + bc;
    }
}

// ---------- beta = softmax(x @ conv_w^T + conv_b) over M+1=3 relations, wave per node ----------
__global__ void __launch_bounds__(256) k_beta(const float* __restrict__ x, const float* __restrict__ conv_w,
                                              const float* __restrict__ conv_b, float* __restrict__ beta) {
    int n = (blockIdx.x << 3) + (threadIdx.x >> 5);      // 50000 waves exactly
    int lane = threadIdx.x & 31;
    const float* xr = x + (size_t)n * FF + lane * 8;
    float4 a0 = *(const float4*)(xr);
    float4 a1 = *(const float4*)(xr + 4);
    float dots[3];
#pragma unroll
    for (int r = 0; r < 3; ++r) {
        const float* wr = conv_w + r * FF + lane * 8;
        float4 w0 = *(const float4*)(wr);
        float4 w1 = *(const float4*)(wr + 4);
        float s = a0.x * w0.x + a0.y * w0.y + a0.z * w0.z + a0.w * w0.w
                + a1.x * w1.x + a1.y * w1.y + a1.z * w1.z + a1.w * w1.w;
#pragma unroll
        for (int off = 16; off; off >>= 1) s += __shfl_xor(s, off, 32);
        dots[r] = s + conv_b[r];
    }
    if (lane == 0) {
        float mx = fmaxf(dots[0], fmaxf(dots[1], dots[2]));
        float e0 = expf(dots[0] - mx), e1 = expf(dots[1] - mx), e2 = expf(dots[2] - mx);
        float inv = 1.0f / (e0 + e1 + e2);
        beta[n * 3 + 0] = e0 * inv;
        beta[n * 3 + 1] = e1 * inv;
        beta[n * 3 + 2] = e2 * inv;
    }
}

// ---------- al/ar: per-node per-head attention logit halves ----------
__global__ void k_attnlogits(const float* __restrict__ h, const float* __restrict__ attn_l,
                             const float* __restrict__ attn_r, float* __restrict__ al, float* __restrict__ ar) {
    int t = blockIdx.x * 256 + threadIdx.x;              // NN*HH threads
    if (t >= NN * HH) return;
    int hh = t & 3;
    const float* hp = h + ((size_t)(t >> 2)) * HD + hh * DD;
    float s0 = 0.f, s1 = 0.f, s2 = 0.f, s3 = 0.f;
#pragma unroll 8
    for (int d = 0; d < DD; ++d) {
        float v = hp[d];
        s0 += v * attn_l[(0 * HH + hh) * DD + d];
        s1 += v * attn_r[(0 * HH + hh) * DD + d];
        s2 += v * attn_l[(1 * HH + hh) * DD + d];
        s3 += v * attn_r[(1 * HH + hh) * DD + d];
    }
    al[0 * NN * HH + t] = s0;  ar[0 * NN * HH + t] = s1;
    al[1 * NN * HH + t] = s2;  ar[1 * NN * HH + t] = s3;
}

// ---------- edge pass 1: segment max via monotone-key atomicMax ----------
__global__ void k_edge_max(const int* __restrict__ edge, const float* __restrict__ al, const float* __restrict__ ar,
                           const float* __restrict__ alpha_act, int mi, unsigned* __restrict__ mkey) {
    int t = blockIdx.x * 256 + threadIdx.x;              // EE*HH threads
    int e = t >> 2, hh = t & 3;
    int src = edge[e], dst = edge[EE + e];
    float lg = (al[src * HH + hh] + ar[dst * HH + hh]) * alpha_act[mi];
    atomicMax(mkey + (size_t)dst * HH + hh, fkey(lg));
}

// ---------- edge pass 2: segment sum of exp ----------
__global__ void k_edge_sum(const int* __restrict__ edge, const float* __restrict__ al, const float* __restrict__ ar,
                           const float* __restrict__ alpha_act, int mi, const unsigned* __restrict__ mkey,
                           float* __restrict__ sbuf) {
    int t = blockIdx.x * 256 + threadIdx.x;              // EE*HH threads
    int e = t >> 2, hh = t & 3;
    int src = edge[e], dst = edge[EE + e];
    float lg = (al[src * HH + hh] + ar[dst * HH + hh]) * alpha_act[mi];
    float w = expf(lg - finv(mkey[(size_t)dst * HH + hh]));
    atomicAdd(sbuf + (size_t)dst * HH + hh, w);
}

// ---------- edge pass 3: weighted message scatter (wave per edge, lane = d) ----------
__global__ void __launch_bounds__(256) k_edge_scatter(const int* __restrict__ edge, const float* __restrict__ al,
                                                      const float* __restrict__ ar, const float* __restrict__ alpha_act,
                                                      int mi, const unsigned* __restrict__ mkey,
                                                      const float* __restrict__ sbuf, const float* __restrict__ h,
                                                      float* __restrict__ rel) {
    int gt = blockIdx.x * 256 + threadIdx.x;
    int e = gt >> 5;                                     // EE waves total
    int lane = gt & 31;
    int src = edge[e], dst = edge[EE + e];
    float wt = 0.f;
    if (lane < 4) {
        float lg = (al[src * HH + lane] + ar[dst * HH + lane]) * alpha_act[mi];
        float mm = finv(mkey[(size_t)dst * HH + lane]);
        float ss = sbuf[(size_t)dst * HH + lane];
        wt = expf(lg - mm) / (ss + 1e-16f) * 0.25f;      // fold mean over H=4 heads
    }
    float w0 = __shfl(wt, 0, 32), w1 = __shfl(wt, 1, 32), w2 = __shfl(wt, 2, 32), w3 = __shfl(wt, 3, 32);
    const float* hp = h + (size_t)src * HD;
    float a0 = hp[lane] * w0 + hp[64 + lane] * w1 + hp[128 + lane] * w2 + hp[192 + lane] * w3;
    float a1 = hp[32 + lane] * w0 + hp[96 + lane] * w1 + hp[160 + lane] * w2 + hp[224 + lane] * w3;
    atomicAdd(rel + (size_t)dst * DD + lane, a0);
    atomicAdd(rel + (size_t)dst * DD + 32 + lane, a1);
}

// ---------- final mix: out = relu(beta0*rel0 + beta1*rel1 + beta2*mean_h) ----------
__global__ void k_final(const float* __restrict__ h, const float* __restrict__ beta,
                        const float* __restrict__ rel0, const float* __restrict__ rel1, float* __restrict__ out) {
    int t = blockIdx.x * 256 + threadIdx.x;              // NN*DD threads exactly
    int n = t >> 6, d = t & 63;
    const float* hp = h + (size_t)n * HD;
    float hm = 0.25f * (hp[d] + hp[64 + d] + hp[128 + d] + hp[192 + d]);
    float v = beta[n * 3 + 0] * rel0[t] + beta[n * 3 + 1] * rel1[t] + beta[n * 3 + 2] * hm;
    out[t] = fmaxf(v, 0.0f);
}

extern "C" void kernel_launch(void* const* d_in, const int* in_sizes, int n_in,
                              void* d_out, int out_size, void* d_ws, size_t ws_size,
                              hipStream_t stream) {
    (void)in_sizes; (void)n_in; (void)out_size; (void)ws_size;
    const float* x         = (const float*)d_in[0];
    const float* W         = (const float*)d_in[1];
    const float* b         = (const float*)d_in[2];
    const float* conv_w    = (const float*)d_in[3];
    const float* conv_b    = (const float*)d_in[4];
    const float* attn_l    = (const float*)d_in[5];
    const float* attn_r    = (const float*)d_in[6];
    const float* alpha_act = (const float*)d_in[7];
    const int*   edge0     = (const int*)d_in[8];
    const int*   edge1     = (const int*)d_in[9];
    float* out = (float*)d_out;

    // carve workspace (~135 MB)
    char* p = (char*)d_ws;
    auto carve = [&](size_t bytes) -> char* {
        char* r = p;
        p += (bytes + 255) & ~(size_t)255;
        return r;
    };
    __bf16*   xhi  = (__bf16*)carve((size_t)NN * FF * 2);
    __bf16*   xlo  = (__bf16*)carve((size_t)NN * FF * 2);
    __bf16*   Bhi  = (__bf16*)carve((size_t)FF * HD * 2);
    __bf16*   Blo  = (__bf16*)carve((size_t)FF * HD * 2);
    float*    h    = (float*)carve((size_t)NN * HD * 4);
    float*    beta = (float*)carve((size_t)NN * 3 * 4);
    float*    al   = (float*)carve((size_t)MM * NN * HH * 4);
    float*    ar   = (float*)carve((size_t)MM * NN * HH * 4);
    unsigned* mkey = (unsigned*)carve((size_t)MM * NN * HH * 4);
    float*    sbuf = (float*)carve((size_t)MM * NN * HH * 4);
    float*    rel  = (float*)carve((size_t)MM * NN * DD * 4);

    // zero accumulators every call (replay-safe)
    hipMemsetAsync(mkey, 0, (size_t)MM * NN * HH * 4, stream);
    hipMemsetAsync(sbuf, 0, (size_t)MM * NN * HH * 4, stream);
    hipMemsetAsync(rel,  0, (size_t)MM * NN * DD * 4, stream);

    // prep + GEMM
    k_convert_x<<<(NN * FF) / 256, 256, 0, stream>>>(x, xhi, xlo);
    k_prep_w<<<(FF * HD) / 256, 256, 0, stream>>>(W, Bhi, Blo);
    k_gemm_h<<<(NN * 16) / 8 / 16 * 16 / 16, 256, 0, stream>>>(xhi, xlo, Bhi, Blo, b, h); // 6250 blocks
    k_beta<<<NN / 8, 256, 0, stream>>>(x, conv_w, conv_b, beta);
    k_attnlogits<<<(NN * HH + 255) / 256, 256, 0, stream>>>(h, attn_l, attn_r, al, ar);

    // edge stages per metapath
    const int* edges[2] = { edge0, edge1 };
    for (int mi = 0; mi < MM; ++mi) {
        const float*  alm = al   + (size_t)mi * NN * HH;
        const float*  arm = ar   + (size_t)mi * NN * HH;
        unsigned*     mk  = mkey + (size_t)mi * NN * HH;
        float*        sb  = sbuf + (size_t)mi * NN * HH;
        float*        rl  = rel  + (size_t)mi * NN * DD;
        k_edge_max<<<(EE * HH) / 256, 256, 0, stream>>>(edges[mi], alm, arm, alpha_act, mi, mk);
        k_edge_sum<<<(EE * HH) / 256, 256, 0, stream>>>(edges[mi], alm, arm, alpha_act, mi, mk, sb);
        k_edge_scatter<<<(EE * 32) / 256, 256, 0, stream>>>(edges[mi], alm, arm, alpha_act, mi, mk, sb, h, rl);
    }

    k_final<<<(NN * DD) / 256, 256, 0, stream>>>(h, beta, rel, rel + (size_t)NN * DD, out);
}
